// CNNBiLSTM_13357348290586
// MI455X (gfx1250) — matrix operand order repaired and checked by reading, not compiled, not run on hardware
//
#include <hip/hip_runtime.h>
#include <hip/hip_bf16.h>
#include <math.h>

// ---------------------------------------------------------------------------
// CNN-BiLSTM for MI455X (gfx1250, wave32).
// Recurrence mapped to v_wmma_f32_16x16x32_f16: per timestep each workgroup
// (16-row batch tile x 1 direction) computes gates(16x128) with 8 waves,
// one 16x16 WMMA N-tile per wave, K=32 == hidden size (K=64 input for layer1
// via two chained WMMAs). h/c state lives in LDS across the 2049-step loop.
// Input fragments are software-pipelined one step ahead; activations use
// raw v_exp_f32 / v_rcp_f32 (no IEEE div fixup chains) to keep the serial
// per-step latency minimal.
// ---------------------------------------------------------------------------

typedef __attribute__((ext_vector_type(16))) _Float16 v16h;
typedef __attribute__((ext_vector_type(8)))  float    v8f;

#define TT   2048          // input time steps
#define TC   2049          // conv output / LSTM time steps
#define BB   256           // batch
#define NBT  16            // batch tiles of 16

// single-instruction v_rcp_f32 (~1 ulp) — avoids div_scale/div_fmas chains
__device__ __forceinline__ float fastrcp(float x) { return __builtin_amdgcn_rcpf(x); }

__device__ __forceinline__ float sigf(float x) {
    return fastrcp(1.0f + __expf(-x));
}
// branch-free tanh: 1 - 2/(1+exp(2x)); exp->{0,inf} gives exact +/-1 saturation
__device__ __forceinline__ float tanhfast(float x) {
    return 1.0f - 2.0f * fastrcp(1.0f + __expf(2.0f * x));
}

// 16-bit A-matrix 16x32 fragment K-map (ISA 7.12.2):
// lanes 0-15: e0..7 -> K0..7,  e8..15 -> K16..23
// lanes16-31: e0..7 -> K8..15, e8..15 -> K24..31
__device__ __forceinline__ int kmapA(int lane, int e) {
    int base = (lane < 16) ? 0 : 8;
    return base + (e < 8 ? e : e + 8);
}

// -------------------------- workspace zero ---------------------------------
__global__ void k_zero(float* __restrict__ p, long n) {
    long i = (long)blockIdx.x * blockDim.x + threadIdx.x;
    if (i < n) p[i] = 0.0f;
}

// ---------------- conv1d(k=2,pad=1) + pack to WMMA A-fragment --------------
// yfrag[bt][t][lane][e] (half), element = conv(b = bt*16 + lane&15, t, h=kmapA)
__global__ void k_conv_pack(const float* __restrict__ x,
                            const float* __restrict__ cw,   // [32][1][2]
                            const float* __restrict__ cb,   // [32]
                            _Float16* __restrict__ yfrag) {
    long idx = (long)blockIdx.x * blockDim.x + threadIdx.x;
    const long N = (long)NBT * TC * 512;
    if (idx >= N) return;
    int e    = (int)(idx & 15);
    int lane = (int)((idx >> 4) & 31);
    long rest = idx >> 9;
    int t  = (int)(rest % TC);
    int bt = (int)(rest / TC);
    int b  = bt * 16 + (lane & 15);
    int h  = kmapA(lane, e);
    float acc = cb[h];
    if (t > 0)  acc += cw[h * 2 + 0] * x[(long)b * TT + (t - 1)];
    if (t < TT) acc += cw[h * 2 + 1] * x[(long)b * TT + t];
    yfrag[idx] = (_Float16)acc;
}

// ------------------------------ LSTM layer 0 -------------------------------
// grid.x = 32: bt = blockIdx.x & 15, dir = blockIdx.x >> 4.  256 thr = 8 waves.
__global__ void __launch_bounds__(256)
k_lstm0(const _Float16* __restrict__ yfrag,
        const float* __restrict__ Wih,    // [2][128][32]
        const float* __restrict__ Whh,    // [2][128][32]
        const float* __restrict__ bias,   // [2][2][128]
        _Float16* __restrict__ h0frag) {  // [2][16][2049][512] half (A-frag)
    const int bt   = blockIdx.x & 15;
    const int dir  = blockIdx.x >> 4;
    const int tid  = threadIdx.x;
    const int lane = tid & 31;
    const int wav  = tid >> 5;
    const int n    = lane & 15;

    __shared__ float    gates[16][128];   // 8 KB
    __shared__ float    cst[16][32];      // 2 KB
    __shared__ _Float16 hst[16][32];      // 1 KB

    for (int i = tid; i < 16 * 32; i += 256) {
        (&cst[0][0])[i] = 0.0f;
        (&hst[0][0])[i] = (_Float16)0.0f;
    }

    // B fragments (32x16 K-major slice of WihT/WhhT), resident for whole loop.
    // B layout: same K-map as A (kmapA): lanes 0-15 K0-7,16-23; lanes 16-31 K8-15,24-31.
    v16h wihB, whhB;
    const int g = wav * 16 + n;
#pragma unroll
    for (int e = 0; e < 16; ++e) {
        int K = kmapA(lane, e);
        wihB[e] = (_Float16)Wih[(long)dir * 128 * 32 + g * 32 + K];
        whhB[e] = (_Float16)Whh[(long)dir * 128 * 32 + g * 32 + K];
    }
    float bsum = bias[dir * 256 + g] + bias[dir * 256 + 128 + g];
    v8f cbias;
#pragma unroll
    for (int r = 0; r < 8; ++r) cbias[r] = bsum;

    __syncthreads();

    const long tilebase = (long)bt * TC;
    // software pipeline: preload fragment for the first timestep
    int t0 = dir ? (TC - 1) : 0;
    v16h xa = *(const v16h*)(yfrag + (tilebase + t0) * 512 + lane * 16);

    for (int tt = 0; tt < TC; ++tt) {
        const int t = dir ? (TC - 1 - tt) : tt;
        v16h ha;
#pragma unroll
        for (int e = 0; e < 16; ++e) ha[e] = hst[n][kmapA(lane, e)];

        v8f acc = cbias;
        acc = __builtin_amdgcn_wmma_f32_16x16x32_f16(false, xa, false, wihB,
                                                     (short)0, acc, false, false);
        acc = __builtin_amdgcn_wmma_f32_16x16x32_f16(false, ha, false, whhB,
                                                     (short)0, acc, false, false);

        // issue next timestep's load now; latency hides behind cell update
        {
            int ttn = (tt + 1 < TC) ? (tt + 1) : tt;       // clamp (last iter dummy)
            int tn  = dir ? (TC - 1 - ttn) : ttn;
            xa = *(const v16h*)(yfrag + (tilebase + tn) * 512 + lane * 16);
        }

        const int mbase = 8 * (lane >> 4);      // C/D layout: M = r + 8*(lane>=16)
#pragma unroll
        for (int r = 0; r < 8; ++r) gates[mbase + r][wav * 16 + n] = acc[r];
        __syncthreads();

#pragma unroll
        for (int rep = 0; rep < 2; ++rep) {
            int idx = tid + rep * 256;
            int m = idx >> 5, j = idx & 31;
            float iv = gates[m][j],      fv = gates[m][32 + j];
            float gv = gates[m][64 + j], ov = gates[m][96 + j];
            float c  = sigf(fv) * cst[m][j] + sigf(iv) * tanhfast(gv);
            float hv = sigf(ov) * tanhfast(c);
            cst[m][j] = c;
            hst[m][j] = (_Float16)hv;
            // scatter into A-fragment order for layer-1 consumption
            int b2 = (j >> 3) & 1;
            int lane2 = m + 16 * b2;
            int e2 = (j & 7) + 8 * ((j >> 4) & 1);
            h0frag[(((long)dir * NBT + bt) * TC + t) * 512 + lane2 * 16 + e2] =
                (_Float16)hv;
        }
        __syncthreads();
    }
}

// ------------------------------ LSTM layer 1 -------------------------------
// Input K=64 (fwd||bwd of layer0) -> two chained WMMAs + recurrent WMMA.
// Cell update fuses attention/FC dot products into per-(b,t) scalars.
__global__ void __launch_bounds__(256)
k_lstm1(const _Float16* __restrict__ h0frag,
        const float* __restrict__ Wih,     // [2][128][64]
        const float* __restrict__ Whh,     // [2][128][32]
        const float* __restrict__ bias,    // [2][2][128]
        const float* __restrict__ attn_w,  // [64]
        const float* __restrict__ fc_w,    // [64]
        float* __restrict__ logit,         // [256][2049]
        float* __restrict__ sdot) {        // [256][2049]
    const int bt   = blockIdx.x & 15;
    const int dir  = blockIdx.x >> 4;
    const int tid  = threadIdx.x;
    const int lane = tid & 31;
    const int wav  = tid >> 5;
    const int n    = lane & 15;

    __shared__ float    gates[16][128];
    __shared__ float    cst[16][32];
    __shared__ _Float16 hst[16][32];

    for (int i = tid; i < 16 * 32; i += 256) {
        (&cst[0][0])[i] = 0.0f;
        (&hst[0][0])[i] = (_Float16)0.0f;
    }

    v16h wiaB, wibB, whhB;
    const int g = wav * 16 + n;
#pragma unroll
    for (int e = 0; e < 16; ++e) {
        int K = kmapA(lane, e);
        wiaB[e] = (_Float16)Wih[(long)dir * 128 * 64 + g * 64 + K];        // K 0..31
        wibB[e] = (_Float16)Wih[(long)dir * 128 * 64 + g * 64 + 32 + K];   // K 32..63
        whhB[e] = (_Float16)Whh[(long)dir * 128 * 32 + g * 32 + K];
    }
    float bsum = bias[dir * 256 + g] + bias[dir * 256 + 128 + g];
    v8f cbias;
#pragma unroll
    for (int r = 0; r < 8; ++r) cbias[r] = bsum;

    __syncthreads();

    const _Float16* basef = h0frag + (((long)0 * NBT + bt) * TC) * 512 + lane * 16;
    const _Float16* baseb = h0frag + (((long)1 * NBT + bt) * TC) * 512 + lane * 16;
    int t0 = dir ? (TC - 1) : 0;
    v16h af = *(const v16h*)(basef + (long)t0 * 512);
    v16h ab = *(const v16h*)(baseb + (long)t0 * 512);

    for (int tt = 0; tt < TC; ++tt) {
        const int t = dir ? (TC - 1 - tt) : tt;
        v16h ha;
#pragma unroll
        for (int e = 0; e < 16; ++e) ha[e] = hst[n][kmapA(lane, e)];

        v8f acc = cbias;
        acc = __builtin_amdgcn_wmma_f32_16x16x32_f16(false, af, false, wiaB,
                                                     (short)0, acc, false, false);
        acc = __builtin_amdgcn_wmma_f32_16x16x32_f16(false, ab, false, wibB,
                                                     (short)0, acc, false, false);
        acc = __builtin_amdgcn_wmma_f32_16x16x32_f16(false, ha, false, whhB,
                                                     (short)0, acc, false, false);

        // pipeline next step's input fragments
        {
            int ttn = (tt + 1 < TC) ? (tt + 1) : tt;
            int tn  = dir ? (TC - 1 - ttn) : ttn;
            af = *(const v16h*)(basef + (long)tn * 512);
            ab = *(const v16h*)(baseb + (long)tn * 512);
        }

        const int mbase = 8 * (lane >> 4);
#pragma unroll
        for (int r = 0; r < 8; ++r) gates[mbase + r][wav * 16 + n] = acc[r];
        __syncthreads();

#pragma unroll
        for (int rep = 0; rep < 2; ++rep) {
            int idx = tid + rep * 256;
            int m = idx >> 5, j = idx & 31;      // wave covers one m row, j = lane
            float iv = gates[m][j],      fv = gates[m][32 + j];
            float gv = gates[m][64 + j], ov = gates[m][96 + j];
            float c  = sigf(fv) * cst[m][j] + sigf(iv) * tanhfast(gv);
            float hv = sigf(ov) * tanhfast(c);
            cst[m][j] = c;
            hst[m][j] = (_Float16)hv;
            // attention logit & fc dot contributions (attn_b cancels in softmax)
            int d = dir * 32 + j;
            float la = hv * attn_w[d];
            float ls = hv * fc_w[d];
#pragma unroll
            for (int off = 16; off; off >>= 1) {     // wave32 reduction over j
                la += __shfl_xor(la, off, 32);
                ls += __shfl_xor(ls, off, 32);
            }
            if (lane == 0) {
                int b = bt * 16 + m;
                atomicAdd(&logit[(long)b * TC + t], la);
                atomicAdd(&sdot [(long)b * TC + t], ls);
            }
        }
        __syncthreads();
    }
}

// --------------------- softmax-pool + FC (scalar form) ---------------------
__global__ void __launch_bounds__(256)
k_attn(const float* __restrict__ logit, const float* __restrict__ sdot,
       const float* __restrict__ fc_b, float* __restrict__ out) {
    const int b = blockIdx.x;
    const int tid = threadIdx.x;
    float m = -INFINITY, s = 0.0f, w = 0.0f;
    for (int t = tid; t < TC; t += 256) {
        float l = logit[(long)b * TC + t];
        float v = sdot [(long)b * TC + t];
        float nm = fmaxf(m, l);
        float e0 = __expf(m - nm), e1 = __expf(l - nm);
        s = s * e0 + e1;
        w = w * e0 + e1 * v;
        m = nm;
    }
    __shared__ float sm[256], ss[256], sw[256];
    sm[tid] = m; ss[tid] = s; sw[tid] = w;
    __syncthreads();
    for (int o = 128; o; o >>= 1) {
        if (tid < o) {
            float m2 = sm[tid + o], s2 = ss[tid + o], w2 = sw[tid + o];
            float nm = fmaxf(sm[tid], m2);
            float e0 = __expf(sm[tid] - nm), e1 = __expf(m2 - nm);
            ss[tid] = ss[tid] * e0 + s2 * e1;
            sw[tid] = sw[tid] * e0 + w2 * e1;
            sm[tid] = nm;
        }
        __syncthreads();
    }
    if (tid == 0) out[b] = sw[0] / ss[0] + fc_b[0];
}

// ---------------------------------------------------------------------------
extern "C" void kernel_launch(void* const* d_in, const int* in_sizes, int n_in,
                              void* d_out, int out_size, void* d_ws, size_t ws_size,
                              hipStream_t stream) {
    (void)in_sizes; (void)n_in; (void)out_size; (void)ws_size;
    const float* x      = (const float*)d_in[0];
    const float* conv_w = (const float*)d_in[1];
    const float* conv_b = (const float*)d_in[2];
    const float* Wih0   = (const float*)d_in[3];
    const float* Whh0   = (const float*)d_in[4];
    const float* b0     = (const float*)d_in[5];
    const float* Wih1   = (const float*)d_in[6];
    const float* Whh1   = (const float*)d_in[7];
    const float* b1     = (const float*)d_in[8];
    const float* attn_w = (const float*)d_in[9];
    /* attn_b (d_in[10]) unused: softmax is shift-invariant */
    const float* fc_w   = (const float*)d_in[11];
    const float* fc_b   = (const float*)d_in[12];
    float* out = (float*)d_out;

    char* ws = (char*)d_ws;
    const size_t YFRAG_B  = (size_t)NBT * TC * 512 * sizeof(_Float16);      // 33.6 MB
    const size_t H0FRAG_B = (size_t)2 * NBT * TC * 512 * sizeof(_Float16);  // 67.1 MB
    _Float16* yfrag  = (_Float16*)ws;
    _Float16* h0frag = (_Float16*)(ws + YFRAG_B);
    float*    logit  = (float*)(ws + YFRAG_B + H0FRAG_B);
    float*    sdot   = logit + (long)BB * TC;

    long nz = 2L * BB * TC;
    k_zero<<<(int)((nz + 255) / 256), 256, 0, stream>>>(logit, nz);

    long np = (long)NBT * TC * 512;
    k_conv_pack<<<(int)((np + 255) / 256), 256, 0, stream>>>(x, conv_w, conv_b, yfrag);

    k_lstm0<<<32, 256, 0, stream>>>(yfrag, Wih0, Whh0, b0, h0frag);
    k_lstm1<<<32, 256, 0, stream>>>(h0frag, Wih1, Whh1, b1, attn_w, fc_w, logit, sdot);
    k_attn<<<BB, 256, 0, stream>>>(logit, sdot, fc_b, out);
}
